// MSiTFAggregation_21801253994522
// MI455X (gfx1250) — compile-verified, matrix operand order checked by reading, not run on hardware
//
#include <hip/hip_runtime.h>
#include <hip/hip_bf16.h>

// ---------------------------------------------------------------------------
// Problem constants (reference: N=16, NVAR=16, L=512, E=768, Q=384)
// ---------------------------------------------------------------------------
#define NB     16
#define MTOK   8192
#define EDIM   768
#define QDIM   384
#define LSEQ   512
#define TBOUND 0.5f
#define EPSG   0.577f
#define LN_DECAY (-0.0030045090202987f)   // ln(0.997)

typedef __bf16  bf16_t;
typedef __attribute__((ext_vector_type(16))) __bf16 v16bf;
typedef __attribute__((ext_vector_type(8)))  float  v8f;
typedef __attribute__((ext_vector_type(4)))  unsigned int u32x4;
typedef __attribute__((ext_vector_type(4)))  int i32x4;
typedef __attribute__((ext_vector_type(8)))  int i32x8;

// LDS row stride (bf16 elems) for the fragment-permuted x tile: 768 + 16 pad
// -> row byte stride 1568 = 392 dwords == 8 mod 64 banks (conflict-free)
#define LDA 784

__device__ __forceinline__ float sigmoidf_(float z) { return 1.0f / (1.0f + __expf(-z)); }

// ===========================================================================
// Kernel W: pre-pack Wk/Wv (f32, [768][384]) -> bf16 in per-lane WMMA
// B-fragment order: packed[((j*24+kb)*2+half)*256 + N*16 + pos]
//   K(pos,half) = (pos&7) + ((pos&8)<<1) + half*8 ;  e = kb*32+K ; col = j*16+N
// ===========================================================================
__global__ void kW(const float* __restrict__ Wk, const float* __restrict__ Wv,
                   bf16_t* __restrict__ wkp, bf16_t* __restrict__ wvp) {
    const int o = blockIdx.x * 256 + threadIdx.x;    // 2 * 294912 total
    const int mat = (o >= EDIM * QDIM) ? 1 : 0;
    const int r = o - mat * (EDIM * QDIM);
    const int pos = r & 15, N = (r >> 4) & 15, half = (r >> 8) & 1, t2 = r >> 9;
    const int kb = t2 % 24, j = t2 / 24;
    const int K = (pos & 7) + ((pos & 8) << 1) + half * 8;
    const int e = kb * 32 + K, col = j * 16 + N;
    const float v = (mat ? Wv : Wk)[(size_t)e * QDIM + col];
    (mat ? wvp : wkp)[r] = (bf16_t)v;
}

// ===========================================================================
// Kernel A: q = query @ Wq + bq  and  qnorm = ||q||
// ===========================================================================
__global__ void kA(const float* __restrict__ query, const float* __restrict__ Wq,
                   const float* __restrict__ bq, float* __restrict__ qvec,
                   float* __restrict__ qnorm) {
    const int n = blockIdx.x, t = threadIdx.x;
    __shared__ float qs[QDIM];
    __shared__ float red[QDIM];
    qs[t] = query[n * QDIM + t];
    __syncthreads();
    float s = bq[t];
    for (int k = 0; k < QDIM; ++k) s += qs[k] * Wq[k * QDIM + t];
    qvec[n * QDIM + t] = s;
    red[t] = s * s;
    __syncthreads();
    if (t < 192) red[t] += red[t + 192];
    __syncthreads();
    if (t < 96) red[t] += red[t + 96];
    __syncthreads();
    if (t == 0) {
        float acc = 0.f;
        for (int i = 0; i < 96; ++i) acc += red[i];
        qnorm[n] = sqrtf(acc);
    }
}

// ===========================================================================
// Kernel B (main WMMA kernel): block = one batch n, 64 m-rows, 8 waves.
//   x tile staged to LDS as bf16 *already in A-fragment order*:
//     xt[row*LDA + kb*32 + half*16 + pos]
//   so one A fragment = one 32-B LDS vector load; B fragments are one 32-B
//   global vector load each from the pre-packed weights.
// ===========================================================================
__global__ void kB(const float* __restrict__ x,
                   const bf16_t* __restrict__ wkp, const float* __restrict__ bk,
                   const bf16_t* __restrict__ wvp, const float* __restrict__ bv,
                   const float* __restrict__ Wcd, const float* __restrict__ bcd,
                   const float* __restrict__ qvec, const float* __restrict__ qnorm,
                   bf16_t* __restrict__ vws, float* __restrict__ attn,
                   float* __restrict__ imp) {
    const int n  = blockIdx.y;
    const int m0 = blockIdx.x * 64;
    const int tid = threadIdx.x;

    extern __shared__ char smem[];
    bf16_t* xt   = (bf16_t*)smem;                 // [64][LDA] bf16, fragment order
    float*  sdot = (float*)(smem + 64 * LDA * 2); // [64]
    float*  snrm = sdot + 64;                     // [64]

    if (tid < 64) { sdot[tid] = 0.f; snrm[tid] = 0.f; }

    // ---- stage x tile into LDS (f32 read, bf16 write in fragment order) ----
    const float* xsrc = x + ((size_t)n * MTOK + m0) * EDIM;
    for (int idx = tid; idx < 64 * EDIM; idx += 256) {
        const int row = idx / EDIM, e = idx - row * EDIM;
        const int kb = e >> 5, r32 = e & 31;
        const int half = (r32 >> 3) & 1;
        const int pos = (r32 & 7) + ((r32 & 16) >> 1);
        xt[row * LDA + kb * 32 + half * 16 + pos] = (bf16_t)xsrc[(size_t)row * EDIM + e];
    }
    __syncthreads();

    // ---- gating path: cd = x @ Wcd + bcd -> importance (4 lanes / row) ----
    {
        const int row = tid >> 2, qt = tid & 3;   // 192 K-values per lane (6 kb blocks)
        float s0 = 0.f, s1 = 0.f;
        // iterate over this lane's quarter of E in fragment-order storage
        for (int eq = 0; eq < 192; ++eq) {
            const int e = qt * 192 + eq;
            const int kb = e >> 5, r32 = e & 31;
            const int half = (r32 >> 3) & 1;
            const int pos = (r32 & 7) + ((r32 & 16) >> 1);
            const float xv = (float)xt[row * LDA + kb * 32 + half * 16 + pos];
            s0 += xv * Wcd[e * 2 + 0];
            s1 += xv * Wcd[e * 2 + 1];
        }
        s0 += __shfl_xor(s0, 1); s0 += __shfl_xor(s0, 2);
        s1 += __shfl_xor(s1, 1); s1 += __shfl_xor(s1, 2);
        if ((tid & 3) == 0) {
            const float p0 = sigmoidf_(s0 + bcd[0]);
            const float p1 = sigmoidf_(s1 + bcd[1]);
            const float l0 = (__logf(p0) + EPSG) * (1.0f / TBOUND);
            const float l1 = (__logf(p1) + EPSG) * (1.0f / TBOUND);
            const float mx = fmaxf(l0, l1);
            const float e0 = __expf(l0 - mx), e1 = __expf(l1 - mx);
            imp[(size_t)n * MTOK + m0 + row] = e1 / (e0 + e1);
        }
    }

    // ---- WMMA GEMM: wave w -> row strip s = w>>1, column parity p = w&1 ----
    const int w    = tid >> 5;
    const int lane = tid & 31;
    const int s    = w >> 1;
    const int p    = w & 1;
    const int half = (lane >= 16) ? 1 : 0;
    const int colN = lane & 15;
    const int Mc   = (lane >= 16) ? 8 : 0;
    const float qn = qnorm[n];

    const bf16_t* arow = xt + (s * 16 + (lane & 15)) * LDA + half * 16;

    float dot8[8], nrm8[8];
#pragma unroll
    for (int r = 0; r < 8; ++r) { dot8[r] = 0.f; nrm8[r] = 0.f; }

    for (int j = p; j < QDIM / 16; j += 2) {
        const int c = j * 16 + colN;
        const float qv  = qvec[n * QDIM + c];
        const float bkc = bk[c], bvc = bv[c];
        v8f acck, accv;
#pragma unroll
        for (int r = 0; r < 8; ++r) { acck[r] = bkc; accv[r] = bvc; }

        const bf16_t* bkrow = wkp + ((size_t)(j * 24) * 2 + half) * 256 + colN * 16;
        const bf16_t* bvrow = wvp + ((size_t)(j * 24) * 2 + half) * 256 + colN * 16;

#pragma unroll 4
        for (int kb = 0; kb < EDIM / 32; ++kb) {
            const v16bf af  = *(const v16bf*)(arow + kb * 32);
            const v16bf bkf = *(const v16bf*)(bkrow + (size_t)kb * 512);
            const v16bf bvf = *(const v16bf*)(bvrow + (size_t)kb * 512);
            acck = __builtin_amdgcn_wmma_f32_16x16x32_bf16(
                false, af, false, bkf, (short)0, acck, false, false);
            accv = __builtin_amdgcn_wmma_f32_16x16x32_bf16(
                false, af, false, bvf, (short)0, accv, false, false);
        }

        // epilogue: spill v (bf16), cosine-sim partials from k
        const int mb = m0 + s * 16;
#pragma unroll
        for (int r = 0; r < 8; ++r) {
            const int M = r + Mc;
            const float kv = acck[r], vv = accv[r];
            vws[((size_t)n * MTOK + mb + M) * QDIM + c] = (bf16_t)vv;
            dot8[r] += kv * qv;
            nrm8[r] += kv * kv;
        }
    }

    // reduce column partials within each 16-lane half, combine via LDS atomics
#pragma unroll
    for (int r = 0; r < 8; ++r) {
        float d = dot8[r], nn = nrm8[r];
        for (int sh = 1; sh < 16; sh <<= 1) { d += __shfl_xor(d, sh); nn += __shfl_xor(nn, sh); }
        if (colN == 0) {
            const int M = s * 16 + r + Mc;
            atomicAdd(&sdot[M], d);
            atomicAdd(&snrm[M], nn);
        }
    }
    __syncthreads();
    if (tid < 64) {
        attn[(size_t)n * MTOK + m0 + tid] = sdot[tid] / (qn * sqrtf(snrm[tid]));
    }
}

// ===========================================================================
// Kernel C: relevance = softmax(attn); retrieval = softmax(rel + .5*imp + .2*rec)
// ===========================================================================
__global__ void kC(const float* __restrict__ attn, const float* __restrict__ imp,
                   float* __restrict__ retr) {
    const int n = blockIdx.x, t = threadIdx.x;
    __shared__ float red[256];
    const float* ap = attn + (size_t)n * MTOK;
    const float* ip = imp  + (size_t)n * MTOK;
    float a[32];
#pragma unroll
    for (int i = 0; i < 32; ++i) a[i] = ap[t + (i << 8)];

    float mx = -1e30f;
#pragma unroll
    for (int i = 0; i < 32; ++i) mx = fmaxf(mx, a[i]);
    red[t] = mx; __syncthreads();
    for (int off = 128; off > 0; off >>= 1) { if (t < off) red[t] = fmaxf(red[t], red[t + off]); __syncthreads(); }
    mx = red[0]; __syncthreads();

    float sum = 0.f;
#pragma unroll
    for (int i = 0; i < 32; ++i) { a[i] = __expf(a[i] - mx); sum += a[i]; }
    red[t] = sum; __syncthreads();
    for (int off = 128; off > 0; off >>= 1) { if (t < off) red[t] += red[t + off]; __syncthreads(); }
    const float inv = 1.0f / red[0]; __syncthreads();

#pragma unroll
    for (int i = 0; i < 32; ++i) {
        const int m = t + (i << 8);
        const float rec = __expf((float)(LSEQ - 1 - (m & (LSEQ - 1))) * LN_DECAY);
        a[i] = a[i] * inv + 0.5f * ip[m] + 0.2f * rec;
    }
    float mx2 = -1e30f;
#pragma unroll
    for (int i = 0; i < 32; ++i) mx2 = fmaxf(mx2, a[i]);
    red[t] = mx2; __syncthreads();
    for (int off = 128; off > 0; off >>= 1) { if (t < off) red[t] = fmaxf(red[t], red[t + off]); __syncthreads(); }
    mx2 = red[0]; __syncthreads();

    float sum2 = 0.f;
#pragma unroll
    for (int i = 0; i < 32; ++i) { a[i] = __expf(a[i] - mx2); sum2 += a[i]; }
    red[t] = sum2; __syncthreads();
    for (int off = 128; off > 0; off >>= 1) { if (t < off) red[t] += red[t + off]; __syncthreads(); }
    const float inv2 = 1.0f / red[0];

    float* rp = retr + (size_t)n * MTOK;
#pragma unroll
    for (int i = 0; i < 32; ++i) rp[t + (i << 8)] = a[i] * inv2;
}

// ===========================================================================
// Kernel D: partial[n,seg,q] = sum_{m in seg} retrieval[n,m] * v[n,m,q]
// Bulk v tile (64x128 bf16) staged via the Tensor Data Mover (TDM 2-D
// descriptor, TENSORcnt); retrieval weights via async global->LDS b128
// (ASYNCcnt). grid = (16, 3, 8), block = 128 (4 waves).
// ===========================================================================
__global__ void kD(const bf16_t* __restrict__ vws, const float* __restrict__ retr,
                   float* __restrict__ partial) {
    const int n = blockIdx.x, qc = blockIdx.y, seg = blockIdx.z, t = threadIdx.x;
    __shared__ __align__(16) bf16_t vs[64 * 128];
    __shared__ __align__(16) float wsh[64];

    const float*  wp  = retr + (size_t)n * MTOK + seg * 1024;
    const bf16_t* vsg = vws + ((size_t)n * MTOK + seg * 1024) * QDIM + qc * 128;
    const unsigned lvs = (unsigned)(unsigned long long)(void*)vs;
    const unsigned lws = (unsigned)(unsigned long long)(void*)wsh;

    float acc = 0.f;
    for (int st = 0; st < 16; ++st) {          // 16 stages x 64 rows = 1024 m
        const int mbase = st * 64;
        if (t < 32) {
            // ---- retrieval weights: async global->LDS (256 B, 16 x b128) ----
            if (t < 16) {
                const unsigned loff = lws + (unsigned)(t * 16);
                const unsigned long long gaw =
                    (unsigned long long)(const void*)(wp + mbase + t * 4);
                asm volatile("global_load_async_to_lds_b128 %0, %1, off"
                             :: "v"(loff), "v"(gaw) : "memory");
            }
#if defined(__has_builtin) && __has_builtin(__builtin_amdgcn_tensor_load_to_lds)
            // ---- v tile: Tensor Data Mover, 2-D tile 128 x 64 (bf16) ----
            const unsigned long long ga =
                (unsigned long long)(const void*)(vsg + (size_t)mbase * QDIM);
            u32x4 g0;
            g0.x = 1u;                                    // count=1 (valid D#)
            g0.y = lvs;                                   // lds_addr
            g0.z = (unsigned)ga;                          // global_addr[31:0]
            g0.w = (unsigned)(ga >> 32) | 0x80000000u;    // addr[56:32] | type=2
            i32x8 g1;
            g1[0] = 0x00010000;          // data_size = 1 (2 bytes)
            g1[1] = (int)(QDIM << 16);   // tensor_dim0 = 384 (bits 79:48)
            g1[2] = (int)(MTOK << 16);   // tensor_dim1 = 8192 (bits 111:80)
            g1[3] = (int)(128 << 16);    // tile_dim0 = 128 (bits 127:112)
            g1[4] = 64;                  // tile_dim1 = 64, tile_dim2 = 0
            g1[5] = QDIM;                // tensor_dim0_stride = 384
            g1[6] = 0;
            g1[7] = 0;
            const i32x4 z4 = {0, 0, 0, 0};
#if __clang_major__ <= 22
            __builtin_amdgcn_tensor_load_to_lds(g0, g1, z4, z4, 0);
#else
            const i32x8 z8 = {0, 0, 0, 0, 0, 0, 0, 0};
            __builtin_amdgcn_tensor_load_to_lds(g0, g1, z4, z4, z8, 0);
#endif
            __builtin_amdgcn_s_wait_tensorcnt(0);
#else
            // ---- fallback: async global->LDS for the v tile ----
            for (int it = 0; it < 32; ++it) {
                const int c = t + it * 32;               // 1024 chunks x 16 B
                const int row = c >> 4, o16 = c & 15;
                const unsigned loff = lvs + (unsigned)(c * 16);
                const unsigned long long gav = (unsigned long long)(const void*)
                    (vsg + (size_t)(mbase + row) * QDIM + o16 * 8);
                asm volatile("global_load_async_to_lds_b128 %0, %1, off"
                             :: "v"(loff), "v"(gav) : "memory");
            }
#endif
            asm volatile("s_wait_asynccnt 0" ::: "memory");
        }
        __syncthreads();
#pragma unroll 8
        for (int r = 0; r < 64; ++r) acc += wsh[r] * (float)vs[r * 128 + t];
        __syncthreads();
    }
    partial[((size_t)n * 8 + seg) * QDIM + qc * 128 + t] = acc;
}

// ===========================================================================
// Kernel E: pooled = sum(partials); out = pooled + Wmu.T@gelu(W1.T@LN(pooled))
// ===========================================================================
__global__ void kE(const float* __restrict__ partial,
                   const float* __restrict__ ln_g, const float* __restrict__ ln_b,
                   const float* __restrict__ W1, const float* __restrict__ b1,
                   const float* __restrict__ Wmu, const float* __restrict__ bmu,
                   float* __restrict__ out) {
    const int n = blockIdx.x, t = threadIdx.x;
    __shared__ float pooled[QDIM], hln[QDIM], h1[4 * QDIM], red[QDIM];

    float s = 0.f;
#pragma unroll
    for (int g = 0; g < 8; ++g) s += partial[((size_t)n * 8 + g) * QDIM + t];
    pooled[t] = s;

    red[t] = s; __syncthreads();
    for (int off = 192; off >= 3; off >>= 1) { if (t < off) red[t] += red[t + off]; __syncthreads(); }
    if (t == 0) red[0] += red[1] + red[2];
    __syncthreads();
    const float mean = red[0] * (1.0f / QDIM);
    __syncthreads();
    const float d = s - mean;
    red[t] = d * d; __syncthreads();
    for (int off = 192; off >= 3; off >>= 1) { if (t < off) red[t] += red[t + off]; __syncthreads(); }
    if (t == 0) red[0] += red[1] + red[2];
    __syncthreads();
    const float rstd = rsqrtf(red[0] * (1.0f / QDIM) + 1e-6f);
    hln[t] = ln_g[t] * d * rstd + ln_b[t];
    __syncthreads();

#pragma unroll
    for (int oi = 0; oi < 4; ++oi) {
        const int o = t + oi * QDIM;
        float acc = b1[o];
        for (int k = 0; k < QDIM; ++k) acc += hln[k] * W1[(size_t)k * (4 * QDIM) + o];
        h1[o] = 0.5f * acc * (1.0f + erff(acc * 0.70710678118f));
    }
    __syncthreads();

    float o2 = bmu[t];
    for (int k = 0; k < 4 * QDIM; ++k) o2 += h1[k] * Wmu[(size_t)k * QDIM + t];
    out[(size_t)n * QDIM + t] = pooled[t] + o2;
}

// ===========================================================================
// Launch
// ===========================================================================
extern "C" void kernel_launch(void* const* d_in, const int* in_sizes, int n_in,
                              void* d_out, int out_size, void* d_ws, size_t ws_size,
                              hipStream_t stream) {
    const float* x     = (const float*)d_in[0];
    const float* query = (const float*)d_in[1];
    const float* Wcd   = (const float*)d_in[2];
    const float* bcd   = (const float*)d_in[3];
    const float* Wk    = (const float*)d_in[4];
    const float* bk    = (const float*)d_in[5];
    const float* Wv    = (const float*)d_in[6];
    const float* bv    = (const float*)d_in[7];
    const float* Wq    = (const float*)d_in[8];
    const float* bq    = (const float*)d_in[9];
    const float* ln_g  = (const float*)d_in[10];
    const float* ln_b  = (const float*)d_in[11];
    const float* W1    = (const float*)d_in[12];
    const float* b1    = (const float*)d_in[13];
    const float* Wmu   = (const float*)d_in[14];
    const float* bmu   = (const float*)d_in[15];
    float* out = (float*)d_out;

    char* w = (char*)d_ws;
    float*  qvec    = (float*)w;   w += (size_t)NB * QDIM * 4;        // 24 KB
    float*  qnorm   = (float*)w;   w += 256;
    float*  attn    = (float*)w;   w += (size_t)NB * MTOK * 4;        // 512 KB
    float*  imp     = (float*)w;   w += (size_t)NB * MTOK * 4;        // 512 KB
    float*  retr    = (float*)w;   w += (size_t)NB * MTOK * 4;        // 512 KB
    float*  partial = (float*)w;   w += (size_t)NB * 8 * QDIM * 4;    // 192 KB
    bf16_t* wkp     = (bf16_t*)w;  w += (size_t)EDIM * QDIM * 2;      // 576 KB
    bf16_t* wvp     = (bf16_t*)w;  w += (size_t)EDIM * QDIM * 2;      // 576 KB
    bf16_t* vws     = (bf16_t*)w;                                     // 100 MB

    kW<<<(2 * EDIM * QDIM) / 256, 256, 0, stream>>>(Wk, Wv, wkp, wvp);
    kA<<<NB, QDIM, 0, stream>>>(query, Wq, bq, qvec, qnorm);

    const size_t smemB = (size_t)64 * LDA * 2 + 512;
    kB<<<dim3(MTOK / 64, NB), 256, smemB, stream>>>(
        x, wkp, bk, wvp, bv, Wcd, bcd, qvec, qnorm, vws, attn, imp);

    kC<<<NB, 256, 0, stream>>>(attn, imp, retr);
    kD<<<dim3(NB, 3, 8), 128, 0, stream>>>(vws, retr, partial);
    kE<<<NB, QDIM, 0, stream>>>(partial, ln_g, ln_b, W1, b1, Wmu, bmu, out);
}
